// ClassicalQuantumAttention_83863531421716
// MI455X (gfx1250) — compile-verified, matrix operand order checked by reading, not run on hardware
//
#include <hip/hip_runtime.h>
#include <hip/hip_fp16.h>

// ---------------------------------------------------------------------------
// Types / WMMA helpers (CDNA5 gfx1250, wave32)
// ---------------------------------------------------------------------------
typedef __attribute__((ext_vector_type(16))) _Float16 v16h;
typedef __attribute__((ext_vector_type(8)))  _Float16 v8h;
typedef __attribute__((ext_vector_type(8)))  float    v8f;

__device__ __forceinline__ v8f wmma_f16(v16h a, v16h b, v8f c) {
  // D = A(16x32,f16) x B(32x16,f16) + C(16x16,f32)
  return __builtin_amdgcn_wmma_f32_16x16x32_f16(
      /*neg_a=*/false, a, /*neg_b=*/false, b,
      /*c_mod=*/(short)0, c, /*reuse_a=*/false, /*reuse_b=*/false);
}

// A/B fragment element -> K index (ISA 7.12.2 16-bit A 16x32 layout):
//   lane group 0 (lanes 0-15):  elems 0-7 -> K0-7,  elems 8-15 -> K16-23
//   lane group 1 (lanes 16-31): elems 0-7 -> K8-15, elems 8-15 -> K24-31
__device__ __forceinline__ int kmap(int e, int lh) {
  return (e & 7) | ((e & 8) << 1) | (lh << 3);
}

// A fragment from a row-major f16 row (stride >= 32, 16B-aligned base):
// elems 0-7 at [base, base+7], elems 8-15 at [base+16, base+23]
// -> two ds_load_b128 / global_load_b128, no per-element packing.
__device__ __forceinline__ v16h load_a16(const _Float16* __restrict__ row, int base) {
  v8h lo = *(const v8h*)(row + base);
  v8h hi = *(const v8h*)(row + base + 16);
  return __builtin_shufflevector(lo, hi, 0, 1, 2, 3, 4, 5, 6, 7,
                                 8, 9, 10, 11, 12, 13, 14, 15);
}

// ---------------------------------------------------------------------------
// Complex helpers for the wave32 state-vector simulator (64 amps = 2/lane)
// amplitude index: bit5 selects register (a0=low half, a1=high half),
// bits 4..0 = lane id.  PennyLane wire w <-> bit (5-w)  (wire 0 = MSB).
// ---------------------------------------------------------------------------
struct C2 { float re, im; };
__device__ __forceinline__ C2 cmul(C2 a, C2 b) {
  return {a.re * b.re - a.im * b.im, a.re * b.im + a.im * b.re};
}
__device__ __forceinline__ C2 cadd(C2 a, C2 b) { return {a.re + b.re, a.im + b.im}; }
__device__ __forceinline__ C2 csel(bool c, C2 a, C2 b) {
  return {c ? a.re : b.re, c ? a.im : b.im};
}
__device__ __forceinline__ C2 shflx(C2 v, int m) {
  return {__shfl_xor(v.re, m, 32), __shfl_xor(v.im, m, 32)};
}
__device__ __forceinline__ float wred(float v) {
#pragma unroll
  for (int o = 16; o > 0; o >>= 1) v += __shfl_xor(v, o, 32);
  return v;
}

__device__ __forceinline__ void apply1q(C2& a0, C2& a1, int wire,
                                        C2 g00, C2 g01, C2 g10, C2 g11, int lane) {
  if (wire == 0) {  // pair is (a0,a1), lane-local
    C2 n0 = cadd(cmul(g00, a0), cmul(g01, a1));
    C2 n1 = cadd(cmul(g10, a0), cmul(g11, a1));
    a0 = n0; a1 = n1;
  } else {          // pair across lanes, butterfly shuffle
    int m = 1 << (5 - wire);
    bool hi = (lane & m) != 0;
    C2 p0 = shflx(a0, m), p1 = shflx(a1, m);
    C2 l0 = cadd(cmul(g00, a0), cmul(g01, p0));
    C2 h0 = cadd(cmul(g10, p0), cmul(g11, a0));
    C2 l1 = cadd(cmul(g00, a1), cmul(g01, p1));
    C2 h1 = cadd(cmul(g10, p1), cmul(g11, a1));
    a0 = csel(hi, h0, l0);
    a1 = csel(hi, h1, l1);
  }
}

__device__ __forceinline__ void rx_gate(C2& a0, C2& a1, int w, float th, int lane) {
  float s, c; __sincosf(0.5f * th, &s, &c);
  apply1q(a0, a1, w, {c, 0.f}, {0.f, -s}, {0.f, -s}, {c, 0.f}, lane);
}
__device__ __forceinline__ void ry_gate(C2& a0, C2& a1, int w, float th, int lane) {
  float s, c; __sincosf(0.5f * th, &s, &c);
  apply1q(a0, a1, w, {c, 0.f}, {-s, 0.f}, {s, 0.f}, {c, 0.f}, lane);
}
__device__ __forceinline__ void rz_gate(C2& a0, C2& a1, int w, float th, int lane) {
  float s, c; __sincosf(0.5f * th, &s, &c);
  apply1q(a0, a1, w, {c, -s}, {0.f, 0.f}, {0.f, 0.f}, {c, s}, lane);
}

// controlled-RX; RX is symmetric so new = cs*a + (-i*sn)*partner for both halves
__device__ __forceinline__ void crx_gate(C2& a0, C2& a1, float th, int c, int t, int lane) {
  float sn, cs; __sincosf(0.5f * th, &sn, &cs);
  if (c == 0) {           // control = bit5: only a1 (bit5==1) is transformed
    int m = 1 << (5 - t);
    C2 p1 = shflx(a1, m);
    a1 = {cs * a1.re + sn * p1.im, cs * a1.im - sn * p1.re};
  } else if (t == 0) {    // target = bit5: lane-local pair, masked by control bit
    int cm = 1 << (5 - c);
    bool ctl = (lane & cm) != 0;
    C2 n0 = {cs * a0.re + sn * a1.im, cs * a0.im - sn * a1.re};
    C2 n1 = {cs * a1.re + sn * a0.im, cs * a1.im - sn * a0.re};
    a0 = csel(ctl, n0, a0);
    a1 = csel(ctl, n1, a1);
  } else {                // both in lane bits
    int cm = 1 << (5 - c), tm = 1 << (5 - t);
    bool ctl = (lane & cm) != 0;
    C2 p0 = shflx(a0, tm), p1 = shflx(a1, tm);
    C2 n0 = {cs * a0.re + sn * p0.im, cs * a0.im - sn * p0.re};
    C2 n1 = {cs * a1.re + sn * p1.im, cs * a1.im - sn * p1.re};
    a0 = csel(ctl, n0, a0);
    a1 = csel(ctl, n1, a1);
  }
}

// one ansatz layer: RX/RY/RZ per qubit, CRX ring forward, CRX ring backward (30 params)
__device__ __forceinline__ void run_layer(C2& a0, C2& a1, const float* __restrict__ p, int lane) {
  int idx = 0;
#pragma unroll
  for (int i = 0; i < 6; ++i) {
    rx_gate(a0, a1, i, p[idx], lane);
    ry_gate(a0, a1, i, p[idx + 1], lane);
    rz_gate(a0, a1, i, p[idx + 2], lane);
    idx += 3;
  }
#pragma unroll
  for (int i = 0; i < 6; ++i) { crx_gate(a0, a1, p[idx++], i, (i + 1) % 6, lane); }
#pragma unroll
  for (int j = 0; j < 6; ++j) {
    int i = 5 - j;
    crx_gate(a0, a1, p[idx++], i, (i == 0) ? 5 : (i - 1), lane);   // python-style mod
  }
}

// ---------------------------------------------------------------------------
// Kernel W: swizzle weights into WMMA fragment-major f16 layout:
//   frag[(ks*NT + nt)*32 + lane][e]  (32B contiguous per lane)
// emb_w:  K=64 (ks 0..1),  N=256 (nt 0..15) -> 32 frags
// att_w1: K=256 (ks 0..7), N=128 (nt 0..7)  -> 64 frags
// proj_w: K=256 (ks 0..7), N=64  (nt 0..3)  -> 32 frags (cols 60..63 zero)
// ---------------------------------------------------------------------------
__global__ __launch_bounds__(128) void prep_kernel(
    const float* __restrict__ emb_w, const float* __restrict__ att_w1,
    const float* __restrict__ proj_w, _Float16* __restrict__ emb_w_f,
    _Float16* __restrict__ att_w1_f, _Float16* __restrict__ proj_w_f) {
  int i = blockIdx.x * 128 + threadIdx.x;  // 0..65535
  int r = i & 511;                          // element within fragment
  int lane = r >> 4, e = r & 15;
  int lh = lane >> 4, lm = lane & 15;
  int k = kmap(e, lh);
  if (i < 16384) {                          // emb_w fragments
    int f = i >> 9, ks = f >> 4, nt = f & 15;
    emb_w_f[i] = (_Float16)emb_w[(ks * 32 + k) * 256 + nt * 16 + lm];
  } else if (i < 49152) {                   // att_w1 fragments
    int j = i - 16384, f = j >> 9, ks = f >> 3, nt = f & 7;
    att_w1_f[j] = (_Float16)att_w1[(ks * 32 + k) * 128 + nt * 16 + lm];
  } else {                                  // proj_w fragments (zero-padded)
    int j = i - 49152, f = j >> 9, ks = f >> 2, nt = f & 3;
    int n = nt * 16 + lm;
    proj_w_f[j] = (n < 60) ? (_Float16)proj_w[(ks * 32 + k) * 60 + n]
                           : (_Float16)0.f;
  }
}

// ---------------------------------------------------------------------------
// Kernel A: fused embed GEMM + attention GEMM + softmax chunk aggregation.
// One block per (b, chunk); 4 waves; x tile read exactly once from HBM,
// weights served from L2 in fragment order (2x b128 per fragment per lane).
// ---------------------------------------------------------------------------
__global__ __launch_bounds__(128) void fused_embed_attn(
    const float* __restrict__ x, const float* __restrict__ emb_b,
    const v16h* __restrict__ emb_w_f, const v16h* __restrict__ att_w1_f,
    const float* __restrict__ att_b1, const float* __restrict__ att_w2,
    const float* __restrict__ att_b2, _Float16* __restrict__ chunks_h) {
  __shared__ __align__(16) _Float16 xT[16][64];       // [t][c] f16, 2KB
  __shared__ float feats[16][256];                    // f32 copy, 16KB
  __shared__ __align__(16) _Float16 feats_h[16][256]; // f16 copy for A-frags, 8KB
  __shared__ float hbuf[16][128];                     // 8KB
  __shared__ float sc[16];
  __shared__ float sw[16];

  const int blk = blockIdx.x;
  const int b = blk >> 7, ch = blk & 127, t0 = ch * 16;
  const int tid = threadIdx.x, wave = tid >> 5, lane = tid & 31;
  const int lh = lane >> 4, lm = lane & 15;

  {  // load & convert 64x16 x tile (4KB fp32), transpose into [t][c]
    int c = tid >> 1, toff = (tid & 1) * 8;
    const float* px = x + ((size_t)b * 64 + c) * 2048 + t0 + toff;
#pragma unroll
    for (int j = 0; j < 8; ++j) xT[toff + j][c] = (_Float16)px[j];
  }
  __syncthreads();

  // feats = xT(16x64) @ emb_w(64x256) + emb_b   -> 8 WMMA per wave
#pragma unroll
  for (int nt = 0; nt < 4; ++nt) {
    const int n0 = wave * 64 + nt * 16;
    const int nf = n0 >> 4;
    v8f acc = {0.f, 0.f, 0.f, 0.f, 0.f, 0.f, 0.f, 0.f};
#pragma unroll
    for (int ks = 0; ks < 2; ++ks) {
      v16h a  = load_a16(&xT[lm][0], ks * 32 + (lh << 3));
      v16h bm = emb_w_f[(size_t)(ks * 16 + nf) * 32 + lane];
      acc = wmma_f16(a, bm, acc);
    }
#pragma unroll
    for (int r = 0; r < 8; ++r) {
      int m = r + (lh << 3);
      float v = acc[r] + emb_b[n0 + lm];
      feats[m][n0 + lm]   = v;
      feats_h[m][n0 + lm] = (_Float16)v;
    }
  }
  __syncthreads();

  // h = tanh(feats(16x256) @ att_w1(256x128) + att_b1)  -> 16 WMMA per wave
#pragma unroll
  for (int nt = 0; nt < 2; ++nt) {
    const int n0 = wave * 32 + nt * 16;
    const int nf = n0 >> 4;
    v8f acc = {0.f, 0.f, 0.f, 0.f, 0.f, 0.f, 0.f, 0.f};
#pragma unroll
    for (int ks = 0; ks < 8; ++ks) {
      v16h a  = load_a16(&feats_h[lm][0], ks * 32 + (lh << 3));
      v16h bm = att_w1_f[(size_t)(ks * 8 + nf) * 32 + lane];
      acc = wmma_f16(a, bm, acc);
    }
#pragma unroll
    for (int r = 0; r < 8; ++r) {
      int m = r + (lh << 3);
      hbuf[m][n0 + lm] = tanhf(acc[r] + att_b1[n0 + lm]);
    }
  }
  __syncthreads();

  // scores (16 per chunk)
  if (tid < 16) {
    float s = att_b2[0];
    for (int n = 0; n < 128; ++n) s += hbuf[tid][n] * att_w2[n];
    sc[tid] = s;
  }
  __syncthreads();

  // softmax weights (computed once by 16 threads)
  if (tid < 16) {
    float mx = sc[0];
#pragma unroll
    for (int t = 1; t < 16; ++t) mx = fmaxf(mx, sc[t]);
    float den = 0.f;
#pragma unroll
    for (int t = 0; t < 16; ++t) den += __expf(sc[t] - mx);
    sw[tid] = __expf(sc[tid] - mx) / den;
  }
  __syncthreads();

  // weighted chunk aggregation -> f16 output row
#pragma unroll
  for (int dd = 0; dd < 2; ++dd) {
    int d = tid + dd * 128;
    float acc2 = 0.f;
#pragma unroll
    for (int t = 0; t < 16; ++t) acc2 += sw[t] * feats[t][d];
    chunks_h[(size_t)blk * 256 + d] = (_Float16)acc2;
  }
}

// ---------------------------------------------------------------------------
// Kernel B: params = sigmoid(chunks @ proj_w + proj_b)   (16384x256 @ 256x64)
// A fragments straight from row-major f16 global (2x b128 per frag).
// ---------------------------------------------------------------------------
__global__ __launch_bounds__(128) void params_kernel(
    const _Float16* __restrict__ chunks_h, const v16h* __restrict__ proj_w_f,
    const float* __restrict__ proj_b, float* __restrict__ params) {
  const int row0 = blockIdx.x * 16;
  const int tid = threadIdx.x, wave = tid >> 5, lane = tid & 31;
  const int lh = lane >> 4, lm = lane & 15;
  const int n0 = wave * 16;   // n-tile == wave
  const _Float16* arow = chunks_h + (size_t)(row0 + lm) * 256 + (lh << 3);
  v8f acc = {0.f, 0.f, 0.f, 0.f, 0.f, 0.f, 0.f, 0.f};
#pragma unroll
  for (int ks = 0; ks < 8; ++ks) {
    v16h a  = load_a16(arow, ks * 32);
    v16h bm = proj_w_f[(size_t)(ks * 4 + wave) * 32 + lane];
    acc = wmma_f16(a, bm, acc);
  }
#pragma unroll
  for (int r = 0; r < 8; ++r) {
    int m = r + (lh << 3), n = n0 + lm;
    float v = acc[r] + (n < 60 ? proj_b[n] : 0.f);
    params[(size_t)(row0 + m) * 64 + n] = 1.f / (1.f + __expf(-v));
  }
}

// ---------------------------------------------------------------------------
// Kernel C: 2-layer ansatz, one wave32 per state (16384 states).
// ---------------------------------------------------------------------------
__global__ __launch_bounds__(256) void ansatz_kernel(
    const float* __restrict__ params, float2* __restrict__ evolved) {
  const int s = blockIdx.x * 8 + (threadIdx.x >> 5);
  const int lane = threadIdx.x & 31;
  const float* p = params + (size_t)s * 64;
  C2 a0 = {lane == 0 ? 1.f : 0.f, 0.f};
  C2 a1 = {0.f, 0.f};
  run_layer(a0, a1, p, lane);
  run_layer(a0, a1, p + 30, lane);
  float2* ev = evolved + (size_t)s * 64;
  ev[lane]      = float2{a0.re, a0.im};
  ev[lane + 32] = float2{a1.re, a1.im};
}

// ---------------------------------------------------------------------------
// Kernel D: LCU mix over chunks + normalize + shared-param ansatz + <X,Y,Z>_i
// One wave per batch element.
// ---------------------------------------------------------------------------
__global__ __launch_bounds__(32) void mix_qff_kernel(
    const float2* __restrict__ evolved, const float* __restrict__ mix_re,
    const float* __restrict__ mix_im, const float* __restrict__ qffp,
    float* __restrict__ qfeat) {
  const int b = blockIdx.x, lane = threadIdx.x;
  float asum = 0.f;
  for (int c = 0; c < 128; ++c)
    asum += sqrtf(mix_re[c] * mix_re[c] + mix_im[c] * mix_im[c]);
  const float cinv = 1.f / (asum + 1e-8f);

  C2 a0 = {0.f, 0.f}, a1 = {0.f, 0.f};
  for (int c = 0; c < 128; ++c) {
    C2 co = {mix_re[c] * cinv, mix_im[c] * cinv};
    const float2* ev = evolved + (size_t)(b * 128 + c) * 64;
    float2 e0 = ev[lane], e1 = ev[lane + 32];
    a0 = cadd(a0, cmul(co, C2{e0.x, e0.y}));
    a1 = cadd(a1, cmul(co, C2{e1.x, e1.y}));
  }
  float nn = wred(a0.re * a0.re + a0.im * a0.im + a1.re * a1.re + a1.im * a1.im);
  float s = 1.f / (sqrtf(nn) + 1e-9f);
  a0.re *= s; a0.im *= s; a1.re *= s; a1.im *= s;

  run_layer(a0, a1, qffp, lane);   // 1 layer, 30 shared params

#pragma unroll
  for (int i = 0; i < 6; ++i) {
    float vx, vy, vz;
    if (i == 0) {  // paulis on bit5: pair is (a0, a1) lane-local
      vx = 2.f * (a0.re * a1.re + a0.im * a1.im);
      vy = 2.f * (a0.re * a1.im - a0.im * a1.re);
      vz = (a0.re * a0.re + a0.im * a0.im) - (a1.re * a1.re + a1.im * a1.im);
    } else {
      int m = 1 << (5 - i);
      C2 p0 = shflx(a0, m), p1 = shflx(a1, m);
      bool hi = (lane & m) != 0;
      vx = a0.re * p0.re + a0.im * p0.im + a1.re * p1.re + a1.im * p1.im;
      float y0 = hi ? (a0.im * p0.re - a0.re * p0.im) : (a0.re * p0.im - a0.im * p0.re);
      float y1 = hi ? (a1.im * p1.re - a1.re * p1.im) : (a1.re * p1.im - a1.im * p1.re);
      vy = y0 + y1;
      float sg = hi ? -1.f : 1.f;
      vz = sg * (a0.re * a0.re + a0.im * a0.im + a1.re * a1.re + a1.im * a1.im);
    }
    vx = wred(vx); vy = wred(vy); vz = wred(vz);
    if (lane == 0) {
      qfeat[b * 18 + i]      = vx;
      qfeat[b * 18 + 6 + i]  = vy;
      qfeat[b * 18 + 12 + i] = vz;
    }
  }
}

// ---------------------------------------------------------------------------
// Kernel E: qfeat @ out_w + LayerNorm + 2-layer classifier head (per b)
// ---------------------------------------------------------------------------
__global__ __launch_bounds__(256) void head_kernel(
    const float* __restrict__ qfeat, const float* __restrict__ out_w,
    const float* __restrict__ out_b, const float* __restrict__ ln_g,
    const float* __restrict__ ln_b, const float* __restrict__ cls_w1,
    const float* __restrict__ cls_b1, const float* __restrict__ cls_w2,
    const float* __restrict__ cls_b2, float* __restrict__ out) {
  __shared__ float q[18];
  __shared__ float obuf[256];
  __shared__ float hbuf[256];
  __shared__ float rbuf[256];
  const int b = blockIdx.x, t = threadIdx.x;
  if (t < 18) q[t] = qfeat[b * 18 + t];
  __syncthreads();

  float v = out_b[t];
#pragma unroll
  for (int k = 0; k < 18; ++k) v += q[k] * out_w[k * 256 + t];

  rbuf[t] = v; __syncthreads();
  for (int s2 = 128; s2 > 0; s2 >>= 1) { if (t < s2) rbuf[t] += rbuf[t + s2]; __syncthreads(); }
  const float mu = rbuf[0] * (1.f / 256.f);
  __syncthreads();
  float dv = v - mu;
  rbuf[t] = dv * dv; __syncthreads();
  for (int s2 = 128; s2 > 0; s2 >>= 1) { if (t < s2) rbuf[t] += rbuf[t + s2]; __syncthreads(); }
  const float var = rbuf[0] * (1.f / 256.f);

  float on = dv * rsqrtf(var + 1e-5f) * ln_g[t] + ln_b[t];
  obuf[t] = on; __syncthreads();

  float hv = cls_b1[t];
  for (int k = 0; k < 256; ++k) hv += obuf[k] * cls_w1[k * 256 + t];
  hbuf[t] = fmaxf(hv, 0.f); __syncthreads();

  if (t < 2) {
    float sacc = cls_b2[t];
    for (int k = 0; k < 256; ++k) sacc += hbuf[k] * cls_w2[k * 2 + t];
    out[b * 2 + t] = sacc;
  }
}

// ---------------------------------------------------------------------------
// Launch
// ---------------------------------------------------------------------------
extern "C" void kernel_launch(void* const* d_in, const int* in_sizes, int n_in,
                              void* d_out, int out_size, void* d_ws, size_t ws_size,
                              hipStream_t stream) {
  (void)in_sizes; (void)n_in; (void)out_size; (void)ws_size;
  const float* x      = (const float*)d_in[0];
  const float* emb_w  = (const float*)d_in[1];
  const float* emb_b  = (const float*)d_in[2];
  const float* att_w1 = (const float*)d_in[3];
  const float* att_b1 = (const float*)d_in[4];
  const float* att_w2 = (const float*)d_in[5];
  const float* att_b2 = (const float*)d_in[6];
  const float* proj_w = (const float*)d_in[7];
  const float* proj_b = (const float*)d_in[8];
  const float* mix_re = (const float*)d_in[9];
  const float* mix_im = (const float*)d_in[10];
  const float* qffp   = (const float*)d_in[11];
  const float* out_w  = (const float*)d_in[12];
  const float* out_b  = (const float*)d_in[13];
  const float* ln_g   = (const float*)d_in[14];
  const float* ln_b   = (const float*)d_in[15];
  const float* cls_w1 = (const float*)d_in[16];
  const float* cls_b1 = (const float*)d_in[17];
  const float* cls_w2 = (const float*)d_in[18];
  const float* cls_b2 = (const float*)d_in[19];
  float* out = (float*)d_out;

  char* ws = (char*)d_ws;
  size_t off = 0;
  auto carve = [&](size_t bytes) -> void* {
    void* r = (void*)(ws + off);
    off += (bytes + 255) & ~(size_t)255;
    return r;
  };
  _Float16* emb_w_f  = (_Float16*)carve((size_t)16384 * 2);          // 32 frags
  _Float16* att_w1_f = (_Float16*)carve((size_t)32768 * 2);          // 64 frags
  _Float16* proj_w_f = (_Float16*)carve((size_t)16384 * 2);          // 32 frags
  _Float16* chunks_h = (_Float16*)carve((size_t)16384 * 256 * 2);
  float*    params   = (float*)carve((size_t)16384 * 64 * 4);
  float2*   evolved  = (float2*)carve((size_t)16384 * 64 * 8);
  float*    qfeat    = (float*)carve((size_t)128 * 18 * 4);

  prep_kernel<<<512, 128, 0, stream>>>(emb_w, att_w1, proj_w,
                                       emb_w_f, att_w1_f, proj_w_f);
  fused_embed_attn<<<16384, 128, 0, stream>>>(x, emb_b, (const v16h*)emb_w_f,
                                              (const v16h*)att_w1_f, att_b1,
                                              att_w2, att_b2, chunks_h);
  params_kernel<<<1024, 128, 0, stream>>>(chunks_h, (const v16h*)proj_w_f,
                                          proj_b, params);
  ansatz_kernel<<<2048, 256, 0, stream>>>(params, evolved);
  mix_qff_kernel<<<128, 32, 0, stream>>>(evolved, mix_re, mix_im, qffp, qfeat);
  head_kernel<<<128, 256, 0, stream>>>(qfeat, out_w, out_b, ln_g, ln_b,
                                       cls_w1, cls_b1, cls_w2, cls_b2, out);
}
